// GAT_23828478558292
// MI455X (gfx1250) — matrix operand order repaired and checked by reading, not compiled, not run on hardware
//
#include <hip/hip_runtime.h>
#include <hip/hip_bf16.h>

// ---------------------------------------------------------------------------
// Types for CDNA5 WMMA (wave32): v_wmma_f32_16x16x32_bf16
// ---------------------------------------------------------------------------
typedef __attribute__((ext_vector_type(16))) __bf16 v16bf;
typedef __attribute__((ext_vector_type(8)))  float  v8f;

union BFrag { v16bf v; uint4 q[2]; };

__device__ __forceinline__ unsigned short f2bf(float f) {
    unsigned int u = __float_as_uint(f);
    unsigned int r = (u + 0x7fffu + ((u >> 16) & 1u)) >> 16;   // RNE
    return (unsigned short)r;
}

__device__ __forceinline__ float lrelu(float e) { return e < 0.f ? 0.2f * e : e; }

__device__ __forceinline__ void atomicMaxF(float* addr, float v) {
    // works for mixed signs when location is initialized to -inf-ish
    if (v >= 0.f) atomicMax((int*)addr, __float_as_int(v));
    else          atomicMin((unsigned int*)addr, (unsigned int)__float_as_int(v));
}

// ---------------------------------------------------------------------------
// fp32 -> bf16 convert (grid stride)
// ---------------------------------------------------------------------------
__global__ void cvt_bf16_kernel(const float* __restrict__ src,
                                unsigned short* __restrict__ dst, int n) {
    for (int i = blockIdx.x * blockDim.x + threadIdx.x; i < n;
         i += gridDim.x * blockDim.x)
        dst[i] = f2bf(src[i]);
}

// W [K,128] fp32 -> Wt [128,K] bf16 (so B fragments read contiguous K)
__global__ void transpose_w_kernel(const float* __restrict__ W,
                                   unsigned short* __restrict__ Wt, int K) {
    int i = blockIdx.x * blockDim.x + threadIdx.x;
    if (i >= K * 128) return;
    int k = i >> 7, c = i & 127;
    Wt[c * K + k] = f2bf(W[i]);
}

__global__ void fill_kernel(float* __restrict__ p, float v, int n) {
    for (int i = blockIdx.x * blockDim.x + threadIdx.x; i < n;
         i += gridDim.x * blockDim.x)
        p[i] = v;
}

// ---------------------------------------------------------------------------
// GEMM: H[N,128] = A[N,K](bf16) * W[K,128](bf16, pre-transposed Wt[128,K])
// 256 threads = 8 waves; wave w -> rows [blk*128 + w*16, +16), all 128 cols.
// Wt staged in LDS via CDNA5 async-copy (GLOBAL_LOAD_ASYNC_TO_LDS_B128,
// ASYNCcnt), then read back as ds_load_b128 B fragments.
// ---------------------------------------------------------------------------
__global__ __launch_bounds__(256) void gemm_bf16_wmma_kernel(
    const unsigned short* __restrict__ A, const unsigned short* __restrict__ Wt,
    float* __restrict__ Hout, int Nn, int K) {
    __shared__ unsigned short lw[128 * 256];   // 64 KB, covers K up to 256

    // --- async stage of Wt (128*K bf16) into LDS, bypassing VGPRs ---------
    // aperture rule: generic LDS address low 32 bits == LDS byte offset
    {
        const int tid = threadIdx.x;
        const int n16 = (128 * K) >> 3;                 // # of 16B chunks
        unsigned lds_base = (unsigned)(unsigned long long)(uintptr_t)(&lw[0]);
        for (int i = tid; i < n16; i += 256) {
            unsigned lds_addr = lds_base + (unsigned)i * 16u;
            unsigned long long ga =
                (unsigned long long)(uintptr_t)(Wt + (size_t)i * 8);
            asm volatile("global_load_async_to_lds_b128 %0, %1, off"
                         :: "v"(lds_addr), "v"(ga) : "memory");
        }
        asm volatile("s_wait_asynccnt 0x0" ::: "memory");
    }
    __syncthreads();

    const int tid  = threadIdx.x;
    const int wave = tid >> 5;
    const int lane = tid & 31;
    const int m16  = lane & 15;          // M (A/B col) within tile
    const int half = lane >> 4;          // K-half selector (contiguous K order used here)
    const int row0 = blockIdx.x * 128 + wave * 16;
    int row  = row0 + m16;
    int rowc = row < Nn ? row : (Nn - 1);
    const unsigned short* aptr = A + (size_t)rowc * K;

    v8f zero = {};
    v8f acc[8];
#pragma unroll
    for (int t = 0; t < 8; ++t) acc[t] = zero;

    for (int k0 = 0; k0 < K; k0 += 32) {
        // A fragment, contiguous K order: elems 0..15 = K(k0+half*16..)
        // (matches B fragment order)
        BFrag a;
        a.q[0] = *(const uint4*)(aptr + k0 + half * 16);
        a.q[1] = *(const uint4*)(aptr + k0 + half * 16 + 8);
        if (k0 + 32 < K) __builtin_prefetch(aptr + k0 + 32, 0, 3);

        // preload all 8 B fragments (common base + immediate t*16*K offsets)
        const unsigned short* bbase = lw + m16 * K + k0 + half * 16;
        BFrag b[8];
#pragma unroll
        for (int t = 0; t < 8; ++t) {
            b[t].q[0] = *(const uint4*)(bbase + t * 16 * K);
            b[t].q[1] = *(const uint4*)(bbase + t * 16 * K + 8);
        }
        // 8 back-to-back WMMAs on the XDL pipe
#pragma unroll
        for (int t = 0; t < 8; ++t)
            acc[t] = __builtin_amdgcn_wmma_f32_16x16x32_bf16(
                false, a.v, false, b[t].v, (short)0, acc[t], false, false);
    }

    // C/D layout: VGPR j -> M=j (lanes 0-15) / M=j+8 (lanes 16-31), N=lane&15
#pragma unroll
    for (int t = 0; t < 8; ++t) {
        int col = t * 16 + m16;
#pragma unroll
        for (int j = 0; j < 8; ++j) {
            int r = row0 + j + 8 * half;
            if (r < Nn) Hout[(size_t)r * 128 + col] = acc[t][j];
        }
    }
}

// ---------------------------------------------------------------------------
// per-node attention logits: esrc[n,h] = <h[n,h,:], a_src[h,:]>, same for dst
// ---------------------------------------------------------------------------
__global__ void attn_logits_kernel(const float* __restrict__ Hh,
                                   const float* __restrict__ a_s,
                                   const float* __restrict__ a_d,
                                   float* __restrict__ esrc,
                                   float* __restrict__ edst, int Nn) {
    int i = blockIdx.x * blockDim.x + threadIdx.x;   // n*4 + h
    if (i >= Nn * 4) return;
    int n = i >> 2, h = i & 3;
    const float* hp = Hh + (size_t)n * 128 + h * 32;
    const float* ap = a_s + h * 32;
    const float* bp = a_d + h * 32;
    float s1 = 0.f, s2 = 0.f;
#pragma unroll
    for (int c = 0; c < 32; ++c) { float v = hp[c]; s1 += v * ap[c]; s2 += v * bp[c]; }
    esrc[i] = s1;
    edst[i] = s2;
}

// ---------------------------------------------------------------------------
// edge pass helpers: edges 0..E-1 from edge_index (int64), E..E+N-1 self loops
// ---------------------------------------------------------------------------
__device__ __forceinline__ void edge_nodes(const long long* __restrict__ ei,
                                           int idx, int E, int Nn, int& s, int& d) {
    if (idx < E) {
        long long ls = ei[idx], ld = ei[(size_t)E + idx];
        s = (int)ls; d = (int)ld;
        s = s < 0 ? 0 : (s >= Nn ? Nn - 1 : s);
        d = d < 0 ? 0 : (d >= Nn ? Nn - 1 : d);
    } else {
        s = d = idx - E;
    }
}

// pass 1: m[d,h] = max over incident edges of leaky_relu(esrc[s]+edst[d])
__global__ void edge_max_kernel(const long long* __restrict__ ei,
                                const float4* __restrict__ esrc,
                                const float4* __restrict__ edst,
                                float* __restrict__ m, int E, int Nn) {
    int idx = blockIdx.x * blockDim.x + threadIdx.x;
    if (idx >= E + Nn) return;
    int s, d; edge_nodes(ei, idx, E, Nn, s, d);
    float4 a = esrc[s], b = edst[d];
    float* mp = m + (size_t)d * 4;
    atomicMaxF(mp + 0, lrelu(a.x + b.x));
    atomicMaxF(mp + 1, lrelu(a.y + b.y));
    atomicMaxF(mp + 2, lrelu(a.z + b.z));
    atomicMaxF(mp + 3, lrelu(a.w + b.w));
}

// pass 2: s[d,h] += exp(e - m[d,h])
__global__ void edge_sum_kernel(const long long* __restrict__ ei,
                                const float4* __restrict__ esrc,
                                const float4* __restrict__ edst,
                                const float4* __restrict__ mv,
                                float* __restrict__ sum, int E, int Nn) {
    int idx = blockIdx.x * blockDim.x + threadIdx.x;
    if (idx >= E + Nn) return;
    int s, d; edge_nodes(ei, idx, E, Nn, s, d);
    float4 a = esrc[s], b = edst[d], mm = mv[d];
    float* sp = sum + (size_t)d * 4;
    atomicAdd(sp + 0, __expf(lrelu(a.x + b.x) - mm.x));
    atomicAdd(sp + 1, __expf(lrelu(a.y + b.y) - mm.y));
    atomicAdd(sp + 2, __expf(lrelu(a.z + b.z) - mm.z));
    atomicAdd(sp + 3, __expf(lrelu(a.w + b.w) - mm.w));
}

// pass 3: one wave32 per edge; lane l handles channels {h*32+l}, h=0..3
__global__ __launch_bounds__(256) void edge_agg_kernel(
    const long long* __restrict__ ei, const float4* __restrict__ esrc,
    const float4* __restrict__ edst, const float4* __restrict__ mv,
    const float4* __restrict__ sv, const float* __restrict__ Hh,
    float* __restrict__ agg, int E, int Nn) {
    int wid  = (blockIdx.x * blockDim.x + threadIdx.x) >> 5;
    int lane = threadIdx.x & 31;
    if (wid >= E + Nn) return;
    int s, d; edge_nodes(ei, wid, E, Nn, s, d);
    float4 a = esrc[s], b = edst[d], mm = mv[d], ss = sv[d];
    float al[4];
    al[0] = __expf(lrelu(a.x + b.x) - mm.x) / ss.x;
    al[1] = __expf(lrelu(a.y + b.y) - mm.y) / ss.y;
    al[2] = __expf(lrelu(a.z + b.z) - mm.z) / ss.z;
    al[3] = __expf(lrelu(a.w + b.w) - mm.w) / ss.w;
    const float* hs = Hh + (size_t)s * 128;
    float* ag = agg + (size_t)d * 128;
#pragma unroll
    for (int t = 0; t < 4; ++t)
        atomicAdd(ag + t * 32 + lane, hs[t * 32 + lane] * al[t]);
}

// out = elu(agg + bias); optional f32 and/or bf16 destinations
__global__ void finalize_kernel(const float* __restrict__ agg,
                                const float* __restrict__ bias,
                                float* __restrict__ outf,
                                unsigned short* __restrict__ outb, int Nn) {
    int i = blockIdx.x * blockDim.x + threadIdx.x;
    if (i >= Nn * 128) return;
    float v = agg[i] + bias[i & 127];
    v = v > 0.f ? v : (__expf(v) - 1.f);
    if (outf) outf[i] = v;
    if (outb) outb[i] = f2bf(v);
}

// ---------------------------------------------------------------------------
// host orchestration
// ---------------------------------------------------------------------------
extern "C" void kernel_launch(void* const* d_in, const int* in_sizes, int n_in,
                              void* d_out, int out_size, void* d_ws, size_t ws_size,
                              hipStream_t stream) {
    const float*     x   = (const float*)d_in[0];
    const long long* ei  = (const long long*)d_in[1];   // int64 [2, E]
    const float*     W1  = (const float*)d_in[2];
    const float*     as1 = (const float*)d_in[3];
    const float*     ad1 = (const float*)d_in[4];
    const float*     b1  = (const float*)d_in[5];
    const float*     W2  = (const float*)d_in[6];
    const float*     as2 = (const float*)d_in[7];
    const float*     ad2 = (const float*)d_in[8];
    const float*     b2  = (const float*)d_in[9];

    const int DIN = 256, HC = 128;
    const int N  = in_sizes[0] / DIN;
    const int E  = in_sizes[1] / 2;
    const int Et = E + N;

    // workspace carve-up (256B aligned)
    char* ws = (char*)d_ws;
    size_t o = 0;
    auto carve = [&](size_t bytes) -> char* {
        o = (o + 255) & ~(size_t)255;
        char* p = ws + o;
        o += bytes;
        return p;
    };
    unsigned short* xb   = (unsigned short*)carve((size_t)N * DIN * 2); // layer1 in (bf16)
    float*          h    = (float*)carve((size_t)N * HC * 4);           // transformed feats
    float*          agg  = (float*)carve((size_t)N * HC * 4);           // aggregation
    float*          esrc = (float*)carve((size_t)N * 4 * 4);
    float*          edst = (float*)carve((size_t)N * 4 * 4);
    float*          m    = (float*)carve((size_t)N * 4 * 4);
    float*          s    = (float*)carve((size_t)N * 4 * 4);
    unsigned short* w1t  = (unsigned short*)carve((size_t)HC * DIN * 2);
    unsigned short* w2t  = (unsigned short*)carve((size_t)HC * HC * 2);
    unsigned short* xb2  = xb;   // layer-2 input aliases xb (xb dead after gemm1)

    const int T = 256;
    // one-time converts
    cvt_bf16_kernel<<<(N * DIN + T - 1) / T, T, 0, stream>>>(x, xb, N * DIN);
    transpose_w_kernel<<<(DIN * HC + T - 1) / T, T, 0, stream>>>(W1, w1t, DIN);
    transpose_w_kernel<<<(HC * HC + T - 1) / T, T, 0, stream>>>(W2, w2t, HC);

    auto layer = [&](const unsigned short* xin, int K, const unsigned short* wt,
                     const float* a_s, const float* a_d, const float* bias,
                     float* outf, unsigned short* outb) {
        gemm_bf16_wmma_kernel<<<(N + 127) / 128, 256, 0, stream>>>(xin, wt, h, N, K);
        attn_logits_kernel<<<(N * 4 + T - 1) / T, T, 0, stream>>>(h, a_s, a_d, esrc, edst, N);
        fill_kernel<<<1024, T, 0, stream>>>(m, -3.0e38f, N * 4);
        fill_kernel<<<1024, T, 0, stream>>>(s, 0.f, N * 4);
        fill_kernel<<<4096, T, 0, stream>>>(agg, 0.f, N * HC);
        edge_max_kernel<<<(Et + T - 1) / T, T, 0, stream>>>(
            ei, (const float4*)esrc, (const float4*)edst, m, E, N);
        edge_sum_kernel<<<(Et + T - 1) / T, T, 0, stream>>>(
            ei, (const float4*)esrc, (const float4*)edst, (const float4*)m, s, E, N);
        edge_agg_kernel<<<((size_t)Et * 32 + T - 1) / T, T, 0, stream>>>(
            ei, (const float4*)esrc, (const float4*)edst, (const float4*)m,
            (const float4*)s, h, agg, E, N);
        finalize_kernel<<<(N * HC + T - 1) / T, T, 0, stream>>>(agg, bias, outf, outb, N);
    };

    // layer 1: bf16 output feeds layer-2 GEMM
    layer(xb, DIN, w1t, as1, ad1, b1, nullptr, xb2);
    // layer 2: f32 output to d_out
    layer(xb2, HC, w2t, as2, ad2, b2, (float*)d_out, nullptr);
}
